// AbsoluteNeuralNetwork_50964081935035
// MI455X (gfx1250) — compile-verified, matrix operand order checked by reading, not run on hardware
//
#include <hip/hip_runtime.h>
#include <math.h>

#define B_DIM 32768
#define D_DIM 1024
#define DEPTH_N 9

typedef __bf16 bf16_t;
typedef bf16_t v16bf __attribute__((ext_vector_type(16)));
typedef float  v8f   __attribute__((ext_vector_type(8)));
typedef unsigned short u16;
typedef unsigned int   u32;
typedef unsigned long long u64;

// ---------- helpers ----------
__device__ __forceinline__ u16 f2bf(float f) {
  union { float f; u32 u; } v; v.f = f;
  u32 r = v.u + 0x7FFFu + ((v.u >> 16) & 1u);   // round-to-nearest-even
  return (u16)(r >> 16);
}

// Async 16B copy global -> LDS (CDNA5, tracked by ASYNCcnt).
// LDS operand = low 32 bits of the flat address (== LDS offset per ISA 10.2).
__device__ __forceinline__ void async_cp16(const u16* g, const u16* l) {
  asm volatile("global_load_async_to_lds_b128 %0, %1, off"
               :: "v"((u32)(size_t)l), "v"((u64)(size_t)g)
               : "memory");
}

// ---------- fp32 -> bf16 (activations), 4 elems/thread ----------
__global__ void cvt_x_kernel(const float* __restrict__ x, u16* __restrict__ o) {
  size_t i = ((size_t)blockIdx.x * blockDim.x + threadIdx.x) * 4;
  float4 v = *(const float4*)(x + i);
  u32 lo = (u32)f2bf(v.x) | ((u32)f2bf(v.y) << 16);
  u32 hi = (u32)f2bf(v.z) | ((u32)f2bf(v.w) << 16);
  *(uint2*)(o + i) = make_uint2(lo, hi);
}

// ---------- fp32 W[k][n] -> bf16 Wt[n][k] (transpose + convert) ----------
__global__ void cvt_w_kernel(const float* __restrict__ w, u16* __restrict__ wt) {
  int idx = blockIdx.x * blockDim.x + threadIdx.x;   // 0 .. D*D-1
  int k = idx >> 10;
  int n = idx & (D_DIM - 1);
  wt[(size_t)n * D_DIM + k] = f2bf(w[idx]);
}

// ---------- abs_out[j] = prod_{d,g<3} cos(aw[d,j,g])^2 / D ----------
__global__ void absvec_kernel(const float* __restrict__ aw, float* __restrict__ av) {
  int j = blockIdx.x * blockDim.x + threadIdx.x;
  if (j >= D_DIM) return;
  float p = 1.0f;
#pragma unroll
  for (int d = 0; d < DEPTH_N; ++d) {
#pragma unroll
    for (int g = 0; g < 3; ++g) {
      float c = cosf(aw[(size_t)d * D_DIM * D_DIM + (size_t)j * D_DIM + g]);
      p *= c * c;
    }
  }
  av[j] = p * (1.0f / (float)D_DIM);
}

// ---------- fused GEMM + bias (+extra) + act, bf16 WMMA f32-accum ----------
// C[m][n] = act( sum_k A[m][k] * Wt[n][k] + bias[n] (+ extra[n]) )
// 256 thr = 8 waves; tile 128x128; K-step 32; wave = 32x64 subtile.
// Double-buffered LDS fed by GLOBAL_LOAD_ASYNC_TO_LDS_B128.
#define TM 128
#define TN 128
#define TK 32
#define NKT (D_DIM / TK)
#define LDSP 48   // padded LDS row stride in u16 (96 B: keeps 16B alignment)

__global__ __launch_bounds__(256)
void gemm_kernel(const u16* __restrict__ A, const u16* __restrict__ Wt,
                 const float* __restrict__ bias, const float* __restrict__ extra,
                 u16* __restrict__ outBf, float* __restrict__ outF, int act)
{
  __shared__ u16 As[2][TM * LDSP];
  __shared__ u16 Bs[2][TN * LDSP];

  const int tid   = threadIdx.x;
  const int lane  = tid & 31;
  const int wave  = tid >> 5;
  const int waveM = wave & 3;     // 4 row groups of 32
  const int waveN = wave >> 2;    // 2 col groups of 64
  const int half  = lane >> 4;    // 0/1
  const int l16   = lane & 15;

  const int m0 = blockIdx.y * TM;
  const int n0 = blockIdx.x * TN;

  // global->LDS mapping: 16B chunks; 4 chunks per 32-elem row; 2 rows/thread
  const int gRow = tid >> 2;            // 0..63
  const int gCol = (tid & 3) * 8;       // 0,8,16,24 (u16 units)
  const u16* aP = A  + (size_t)(m0 + gRow) * D_DIM + gCol;
  const u16* bP = Wt + (size_t)(n0 + gRow) * D_DIM + gCol;

  v8f acc[2][4];
#pragma unroll
  for (int mt = 0; mt < 2; ++mt)
#pragma unroll
    for (int nt = 0; nt < 4; ++nt)
      acc[mt][nt] = (v8f){0.f,0.f,0.f,0.f,0.f,0.f,0.f,0.f};

  union Frag { v16bf v; uint4 q[2]; };

  auto issue = [&](int buf, int kt) {
    async_cp16(aP + kt,                       &As[buf][ gRow       * LDSP + gCol]);
    async_cp16(aP + kt + (size_t)64 * D_DIM,  &As[buf][(gRow + 64) * LDSP + gCol]);
    async_cp16(bP + kt,                       &Bs[buf][ gRow       * LDSP + gCol]);
    async_cp16(bP + kt + (size_t)64 * D_DIM,  &Bs[buf][(gRow + 64) * LDSP + gCol]);
  };

  issue(0, 0);                                   // prologue: stage 0 in flight

  for (int i = 0; i < NKT; ++i) {
    const int cur = i & 1;

    // my stage-i async loads are done (they overlapped stage i-1's math)
    asm volatile("s_wait_asynccnt 0x0" ::: "memory");
    __syncthreads();   // all waves' loads visible; all stage i-1 LDS reads retired

    if (i + 1 < NKT) {
      issue(cur ^ 1, (i + 1) * TK);              // next stage into other buffer
      if (i + 2 < NKT) {                          // nudge stage i+2 toward L2
        __builtin_prefetch(aP + (i + 2) * TK, 0, 1);
        __builtin_prefetch(bP + (i + 2) * TK, 0, 1);
      }
    }

    Frag afr[2], bfr[4];
#pragma unroll
    for (int mt = 0; mt < 2; ++mt) {
      int m = waveM * 32 + mt * 16 + l16;
      afr[mt].q[0] = *(const uint4*)&As[cur][m * LDSP + half * 8];       // K 0..15 half
      afr[mt].q[1] = *(const uint4*)&As[cur][m * LDSP + 16 + half * 8];  // K 16..31 half
    }
#pragma unroll
    for (int nt = 0; nt < 4; ++nt) {
      int n = waveN * 64 + nt * 16 + l16;
      bfr[nt].q[0] = *(const uint4*)&Bs[cur][n * LDSP + half * 8];
      bfr[nt].q[1] = *(const uint4*)&Bs[cur][n * LDSP + 16 + half * 8];
    }
#pragma unroll
    for (int mt = 0; mt < 2; ++mt)
#pragma unroll
      for (int nt = 0; nt < 4; ++nt)
        acc[mt][nt] = __builtin_amdgcn_wmma_f32_16x16x32_bf16(
            false, afr[mt].v, false, bfr[nt].v,
            (short)0, acc[mt][nt], false, false);
  }

  // epilogue: C/D layout -> lane gives N = l16, VGPR r gives M = r + half*8
#pragma unroll
  for (int nt = 0; nt < 4; ++nt) {
    int n = n0 + waveN * 64 + nt * 16 + l16;
    float bn = bias[n];
    if (extra) bn += extra[n];
#pragma unroll
    for (int mt = 0; mt < 2; ++mt) {
#pragma unroll
      for (int r = 0; r < 8; ++r) {
        int m = m0 + waveM * 32 + mt * 16 + half * 8 + r;
        float v = acc[mt][nt][r] + bn;
        v = (act == 0) ? fmaxf(v, 0.0f) : tanhf(v);
        if (outBf) outBf[(size_t)m * D_DIM + n] = f2bf(v);
        if (outF)  outF [(size_t)m * D_DIM + n] = v;
      }
    }
  }
}

// ---------- driver ----------
extern "C" void kernel_launch(void* const* d_in, const int* in_sizes, int n_in,
                              void* d_out, int out_size, void* d_ws, size_t ws_size,
                              hipStream_t stream) {
  const float* x   = (const float*)d_in[0];
  const float* w0  = (const float*)d_in[1];
  const float* b0  = (const float*)d_in[2];
  const float* w1  = (const float*)d_in[3];
  const float* b1  = (const float*)d_in[4];
  const float* aw1 = (const float*)d_in[5];
  const float* cw1 = (const float*)d_in[6];
  const float* cb1 = (const float*)d_in[7];
  const float* w2  = (const float*)d_in[8];
  const float* b2  = (const float*)d_in[9];
  const float* aw2 = (const float*)d_in[10];
  const float* cw2 = (const float*)d_in[11];
  const float* cb2 = (const float*)d_in[12];
  const float* w3  = (const float*)d_in[13];
  const float* b3  = (const float*)d_in[14];
  float* out = (float*)d_out;

  char* ws = (char*)d_ws;
  const size_t actBytes = (size_t)B_DIM * D_DIM * sizeof(u16);   // 64 MB
  const size_t wElems   = (size_t)D_DIM * D_DIM;
  u16* actA = (u16*)ws;
  u16* actB = (u16*)(ws + actBytes);
  u16* wtBase = (u16*)(ws + 2 * actBytes);
  u16* wt0 = wtBase + 0 * wElems;
  u16* wt1 = wtBase + 1 * wElems;
  u16* wt2 = wtBase + 2 * wElems;   // cw1
  u16* wt3 = wtBase + 3 * wElems;   // w2
  u16* wt4 = wtBase + 4 * wElems;   // cw2
  u16* wt5 = wtBase + 5 * wElems;   // w3
  float* av1 = (float*)(wtBase + 6 * wElems);
  float* av2 = av1 + D_DIM;

  // pre-passes
  cvt_x_kernel<<<(B_DIM * (size_t)D_DIM) / 4 / 256, 256, 0, stream>>>(x, actA);
  const int wGrid = (D_DIM * D_DIM) / 256;
  cvt_w_kernel<<<wGrid, 256, 0, stream>>>(w0,  wt0);
  cvt_w_kernel<<<wGrid, 256, 0, stream>>>(w1,  wt1);
  cvt_w_kernel<<<wGrid, 256, 0, stream>>>(cw1, wt2);
  cvt_w_kernel<<<wGrid, 256, 0, stream>>>(w2,  wt3);
  cvt_w_kernel<<<wGrid, 256, 0, stream>>>(cw2, wt4);
  cvt_w_kernel<<<wGrid, 256, 0, stream>>>(w3,  wt5);
  absvec_kernel<<<D_DIM / 256, 256, 0, stream>>>(aw1, av1);
  absvec_kernel<<<D_DIM / 256, 256, 0, stream>>>(aw2, av2);

  // six fused GEMM layers (ping-pong bf16 activations)
  dim3 grid(D_DIM / TN, B_DIM / TM);
  gemm_kernel<<<grid, 256, 0, stream>>>(actA, wt0, b0,  nullptr, actB, nullptr, 0); // relu
  gemm_kernel<<<grid, 256, 0, stream>>>(actB, wt1, b1,  nullptr, actA, nullptr, 0); // relu
  gemm_kernel<<<grid, 256, 0, stream>>>(actA, wt2, cb1, av1,     actB, nullptr, 1); // tanh
  gemm_kernel<<<grid, 256, 0, stream>>>(actB, wt3, b2,  nullptr, actA, nullptr, 0); // relu
  gemm_kernel<<<grid, 256, 0, stream>>>(actA, wt4, cb2, av2,     actB, nullptr, 1); // tanh
  gemm_kernel<<<grid, 256, 0, stream>>>(actB, wt5, b3,  nullptr, nullptr, out,   0); // relu -> fp32
}
